// DictionaryLearning_24850680774739
// MI455X (gfx1250) — matrix operand order repaired and checked by reading, not compiled, not run on hardware
//
#include <hip/hip_runtime.h>
#include <hip/hip_bf16.h>

// ---------------------------------------------------------------------------
// VQ dictionary-learning (batched OMP, sparsity 5) for MI455X / gfx1250.
//   X  = tokens [64 x 32768] from z_e[32,64,32,32]  (signal s = l*32 + b)
//   Dn = unit-norm atoms [64 x 512]
// Heavy op: corr = Dn^T @ residual  -> v_wmma_f32_16x16x32_bf16 (selection),
// exact f32 for alpha / residual / outputs (z_dl = X - residual_final,
// loss = 1.25 * mean(residual_final^2)).
// All WMMAs use independent zero-seeded accumulators (no D->C chains, so no
// WMMA->WMMA hazard padding); K-half sum is folded into the branchless
// packed-key argmax scan, which also fills the WMMA->VALU hazard windows.
// ---------------------------------------------------------------------------

typedef __attribute__((ext_vector_type(16))) __bf16 v16bf;
typedef __attribute__((ext_vector_type(8)))  __bf16 v8bf;
typedef __attribute__((ext_vector_type(8)))  float  v8f;
typedef __attribute__((ext_vector_type(4)))  float  v4f;

#define NEMB      512
#define CDIM      64
#define NSIG      32768
#define SIGPB     128          // signals per workgroup (8 waves * 16)
#define SPARSITY  5
#define DPAD      72           // bf16 row pad: 144B rows -> 16B aligned chunks
#define RPAD      72           // f32  row pad: 288B rows -> 16B aligned chunks
#define ZDL_ELEMS (32 * 64 * 1024)      // 2097152
#define LOSS_OFF  ZDL_ELEMS
#define COEFF_OFF (ZDL_ELEMS + 1)
#define COEFF_ELEMS ((size_t)NEMB * NSIG)

// --- prep: normalize atoms; write f32 column-major DnT[512][64] and bf16 copy
__global__ __launch_bounds__(256) void dl_prep_kernel(const float* __restrict__ D,
                                                      float* __restrict__ DnT,
                                                      __bf16* __restrict__ DTb) {
  int n = blockIdx.x * blockDim.x + threadIdx.x;
  if (n >= NEMB) return;
  float col[CDIM];
  float ss = 0.f;
#pragma unroll
  for (int c = 0; c < CDIM; ++c) {
    float v = D[c * NEMB + n];
    col[c] = v;
    ss += v * v;
  }
  float inv = 1.f / fmaxf(sqrtf(ss), 1e-10f);
#pragma unroll
  for (int c = 0; c < CDIM; ++c) {
    float v = col[c] * inv;
    DnT[n * CDIM + c] = v;          // column of Dn, contiguous
    DTb[n * CDIM + c] = (__bf16)v;  // bf16 for WMMA A operand
  }
}

// --- init: zero loss slot + coefficient region (d_out is poisoned 0xAA)
__global__ __launch_bounds__(256) void dl_zero_kernel(float* __restrict__ out) {
  size_t stride = (size_t)gridDim.x * blockDim.x;
  size_t total = COEFF_ELEMS + 1;   // loss + coeffs, contiguous from LOSS_OFF
  for (size_t i = (size_t)blockIdx.x * blockDim.x + threadIdx.x; i < total; i += stride)
    out[LOSS_OFF + i] = 0.f;
}

// --- main OMP kernel: one workgroup = 128 signals, one wave = 16 signals
__global__ __launch_bounds__(256) void dl_omp_kernel(const float* __restrict__ z_e,
                                                     const float* __restrict__ DnT,
                                                     const __bf16* __restrict__ DTb,
                                                     float* __restrict__ out) {
  __shared__ __bf16 sDT[NEMB][DPAD];   // transposed bf16 dictionary (all waves)
  __shared__ float  sR[SIGPB][RPAD];   // f32 residual, [signal][channel]

  const int tid = threadIdx.x;
  const int s0  = blockIdx.x * SIGPB;

  // Load bf16 dictionary into LDS (rows 16B-aligned thanks to DPAD=72).
  for (int i = tid; i < NEMB * CDIM; i += 256) {
    int a = i >> 6, c = i & 63;
    sDT[a][c] = DTb[i];
  }
  // Load residual = tokens. token[s][c] = z_e[b*65536 + c*1024 + l], s = l*32+b.
  for (int i = tid; i < SIGPB * CDIM; i += 256) {
    int j = i >> 6, c = i & 63;
    int s = s0 + j, b = s & 31, l = s >> 5;
    sR[j][c] = z_e[(size_t)b * 65536 + (size_t)c * 1024 + l];
  }
  __syncthreads();
  // After this point sDT is read-only and each wave touches only its own 16
  // rows of sR (half-lanes partition the channels) -> no more barriers needed.

  const int lane = tid & 31;
  const int hl   = lane & 15;     // signal slot within wave
  const int hi   = lane >> 4;     // half-lane selector
  const int j    = ((tid >> 5) << 4) + hl;   // local signal index
  const int sg   = s0 + j;                   // global signal index

  // kill[r] bit t == 1  -> atom (t*16 + hi*8 + r) already selected (masked)
  unsigned int kill[8] = {0u, 0u, 0u, 0u, 0u, 0u, 0u, 0u};

  // A-fragment loader: row M = hl -> atom n0+hl.
  // lanes 0-15: K = 0..15 ; lanes 16-31: K = 16..31.
  auto loadA = [&](int n0, v16bf& alo, v16bf& ahi) {
    const v8bf* prow = (const v8bf*)&sDT[n0 + hl][0];   // 8 chunks of 8 bf16
    v8bf c01 = prow[2 * hi],     c23 = prow[2 * hi + 1];
    v8bf c45 = prow[4 + 2 * hi], c67 = prow[5 + 2 * hi];
    alo = __builtin_shufflevector(c01, c23, 0,1,2,3,4,5,6,7,8,9,10,11,12,13,14,15);
    ahi = __builtin_shufflevector(c45, c67, 0,1,2,3,4,5,6,7,8,9,10,11,12,13,14,15);
  };

  for (int step = 0; step < SPARSITY; ++step) {
    // ---- B fragments (residual, bf16) for K=0..31 (blo) and K=32..63 (bhi).
    // Layout: lanes 0-15 hold K=e, lanes 16-31 hold K=16+e, N = hl.
    const int kb = hi * 16;
    const v4f* rrow = (const v4f*)&sR[j][0];    // 16B-aligned chunks of 4 f32
    v4f r0 = rrow[kb / 4 + 0], r1 = rrow[kb / 4 + 1];
    v4f r2 = rrow[kb / 4 + 2], r3 = rrow[kb / 4 + 3];
    v4f r4 = rrow[8 + kb / 4 + 0], r5 = rrow[8 + kb / 4 + 1];
    v4f r6 = rrow[8 + kb / 4 + 2], r7 = rrow[8 + kb / 4 + 3];
    v16bf blo, bhi;
#pragma unroll
    for (int e = 0; e < 4; ++e) {
      blo[e]      = (__bf16)r0[e];
      blo[4 + e]  = (__bf16)r1[e];
      blo[8 + e]  = (__bf16)r2[e];
      blo[12 + e] = (__bf16)r3[e];
      bhi[e]      = (__bf16)r4[e];
      bhi[4 + e]  = (__bf16)r5[e];
      bhi[8 + e]  = (__bf16)r6[e];
      bhi[12 + e] = (__bf16)r7[e];
    }

    unsigned int bestKey = 0u;   // (|corr| bits & ~511) | (511 - atom)

    // Branchless masked argmax of one 16-atom tile: corr[r] = p[r] + q[r]
    // (the two K-half accumulators), key packs (|corr|, atom), kill mask is
    // a sign-extend bitwise clear (no VCC).
    auto scanTile = [&](const v8f& p, const v8f& q, int n0s) {
      const int          sh    = 31 - (n0s >> 4);                 // SGPR shift
      const unsigned int abase = (unsigned int)(511 - n0s - hi * 8);
#pragma unroll
      for (int r = 0; r < 8; ++r) {
        float corr = p[r] + q[r];
        unsigned int bits = (__float_as_uint(corr) & 0x7FFFFE00u) | (abase - r);
        unsigned int m    = (unsigned int)((int)(kill[r] << sh) >> 31);
        unsigned int key  = bits & ~m;
        bestKey = bestKey > key ? bestKey : key;
      }
    };

    // ---- 32 tiles, unrolled x2. All four WMMAs are independent (separate
    // zero-seeded accumulators, no D->C chains): no WMMA->WMMA hazards, and
    // scan0's VALU separates tile 1's WMMAs from scan1.
    for (int n0 = 0; n0 < NEMB; n0 += 32) {
      v16bf a0lo, a0hi, a1lo, a1hi;
      loadA(n0, a0lo, a0hi);
      loadA(n0 + 16, a1lo, a1hi);
      v8f p0 = {}, q0 = {}, p1 = {}, q1 = {};
      p0 = __builtin_amdgcn_wmma_f32_16x16x32_bf16(false, a0lo, false, blo,
                                                   (short)0, p0, false, false);
      q0 = __builtin_amdgcn_wmma_f32_16x16x32_bf16(false, a0hi, false, bhi,
                                                   (short)0, q0, false, false);
      p1 = __builtin_amdgcn_wmma_f32_16x16x32_bf16(false, a1lo, false, blo,
                                                   (short)0, p1, false, false);
      q1 = __builtin_amdgcn_wmma_f32_16x16x32_bf16(false, a1hi, false, bhi,
                                                   (short)0, q1, false, false);
      scanTile(p0, q0, n0);
      scanTile(p1, q1, n0 + 16);
    }

    // ---- combine the two half-lane candidates (wave32 xor-16 shuffle)
    unsigned int oKey = (unsigned int)__shfl_xor((int)bestKey, 16, 32);
    bestKey = bestKey > oKey ? bestKey : oKey;
    const int idx = 511 - (int)(bestKey & 511u);

    // ---- mark idx as selected in this lane's kill bitmap (branchless)
#pragma unroll
    for (int r = 0; r < 8; ++r) {
      bool mine = ((idx & 7) == r) & (((idx >> 3) & 1) == hi);
      kill[r] |= mine ? (1u << (idx >> 4)) : 0u;
    }

    // ---- exact f32 alpha: lane handles 32 channels of its signal
    const float* dcol = DnT + (size_t)idx * CDIM;
    const int cb = hi * 32;
    float num = 0.f, den = 0.f;
#pragma unroll
    for (int c = 0; c < 32; ++c) {
      float d = dcol[cb + c];
      float rr = sR[j][cb + c];
      num = fmaf(rr, d, num);
      den = fmaf(d, d, den);
    }
    num += __shfl_xor(num, 16, 32);
    den += __shfl_xor(den, 16, 32);
    const float alpha = num / (den + 1e-10f);

    // ---- residual update (exact f32)
#pragma unroll
    for (int c = 0; c < 32; ++c)
      sR[j][cb + c] -= alpha * dcol[cb + c];

    if (hi == 0)
      out[COEFF_OFF + (size_t)idx * NSIG + sg] = alpha;
  }

  // ---- outputs: z_dl = x - residual ; loss = 1.25 * mean(residual^2)
  // sR rows are wave-private, so no barrier is required here either.
  float lsum = 0.f;
  for (int i = lane; i < 16 * CDIM; i += 32) {
    int jj = ((tid >> 5) << 4) + (i >> 6), c = i & 63;
    int s = s0 + jj, b = s & 31, l = s >> 5;
    size_t g = (size_t)b * 65536 + (size_t)c * 1024 + l;
    float r = sR[jj][c];
    out[g] = z_e[g] - r;
    lsum = fmaf(r, r, lsum);
  }
#pragma unroll
  for (int off = 16; off > 0; off >>= 1)
    lsum += __shfl_xor(lsum, off, 32);
  if (lane == 0)
    atomicAdd(out + LOSS_OFF, lsum * (1.25f / (float)ZDL_ELEMS));
}

extern "C" void kernel_launch(void* const* d_in, const int* in_sizes, int n_in,
                              void* d_out, int out_size, void* d_ws, size_t ws_size,
                              hipStream_t stream) {
  const float* z_e = (const float*)d_in[0];   // [32,64,32,32]
  const float* D   = (const float*)d_in[1];   // [64,512]
  float* out = (float*)d_out;                 // [z_dl | loss | coeffs]

  float*  DnT = (float*)d_ws;                                  // 512*64 f32
  __bf16* DTb = (__bf16*)((char*)d_ws + NEMB * CDIM * 4);      // 512*64 bf16

  dl_prep_kernel<<<(NEMB + 255) / 256, 256, 0, stream>>>(D, DnT, DTb);
  dl_zero_kernel<<<4096, 256, 0, stream>>>(out);
  dl_omp_kernel<<<NSIG / SIGPB, 256, 0, stream>>>(z_e, DnT, DTb, out);
}
